// AtomBlock_35983236006592
// MI455X (gfx1250) — matrix operand order repaired and checked by reading, not compiled, not run on hardware
//
#include <hip/hip_runtime.h>
#include <hip/hip_bf16.h>
#include <math.h>

#define D_ATOM 128
#define D_MODEL 512
#define NHEAD 4
#define D_H 32
#define WIN 16
#define WSPAN 33
#define ATT_BI 32   // atoms per attention block

typedef __attribute__((ext_vector_type(16))) __bf16 v16bf;
typedef __attribute__((ext_vector_type(8)))  __bf16 v8bf;
typedef __attribute__((ext_vector_type(8)))  float  v8f;

__device__ __forceinline__ float sigmoidf_(float x) { return 1.0f / (1.0f + __expf(-x)); }

// ---------------------------------------------------------------------------
// WMMA GEMM: C[M,N] = gather(A)[M,K] @ W[N,K]^T (+ bias[N]); A,W bf16, acc f32.
// Wave tile = 32(M) x 64(N): 2 A-fragments x 4 B-fragments -> 8 WMMAs per
// 32-wide K step against 8 fragment loads. bf16 operands load directly in the
// same contiguous k = 16*(lane>>4) + s order (A: two contiguous 8-half chunks; B: 16 contiguous
// halves of a weight row). Output stored f32 or bf16 (bf16 when the consumer
// is another GEMM).
// ---------------------------------------------------------------------------
__global__ __launch_bounds__(256) void gemm_bf16_wmma_kernel(
    const __bf16* __restrict__ A, const int* __restrict__ rowidx,
    const __bf16* __restrict__ W, const float* __restrict__ bias,
    float* __restrict__ Cf, __bf16* __restrict__ Cb,
    int M, int N, int K)
{
  const int lane  = threadIdx.x & 31;
  const int wave  = threadIdx.x >> 5;
  const int tilesN = N >> 6;
  const int wt = blockIdx.x * 8 + wave;
  const int tM = wt / tilesN;          // 32-row tile index
  const int tN = wt % tilesN;          // 64-col tile index
  if ((tM << 5) >= M) return;          // wave-uniform

  const int m0 = (tM << 5) + (lane & 15);
  const int m1 = m0 + 16;
  const __bf16* arow0 = A + (size_t)(rowidx ? rowidx[m0] : m0) * K;
  const __bf16* arow1 = A + (size_t)(rowidx ? rowidx[m1] : m1) * K;
  const int aoff  = (lane & 16) ? 16 : 0;   // A: lanes 16-31 hold K+16 block
  const int nb    = (tN << 6) + (lane & 15);
  const int kboff = (lane & 16) ? 16 : 0;   // B: lanes 16-31 hold K+16 block

  v8f acc[2][4] = {};

  for (int k0 = 0; k0 < K; k0 += 32) {
    // prefetch next K chunk of the A rows (global_prefetch_b8)
    if (k0 + 32 < K) {
      __builtin_prefetch((const void*)(arow0 + k0 + 32 + aoff), 0, 0);
      __builtin_prefetch((const void*)(arow1 + k0 + 32 + aoff), 0, 0);
    }
    const v8bf a0lo = *(const v8bf*)(arow0 + k0 + aoff);
    const v8bf a0hi = *(const v8bf*)(arow0 + k0 + aoff + 8);
    const v8bf a1lo = *(const v8bf*)(arow1 + k0 + aoff);
    const v8bf a1hi = *(const v8bf*)(arow1 + k0 + aoff + 8);
    const v16bf a0 = __builtin_shufflevector(a0lo, a0hi,
        0,1,2,3,4,5,6,7,8,9,10,11,12,13,14,15);
    const v16bf a1 = __builtin_shufflevector(a1lo, a1hi,
        0,1,2,3,4,5,6,7,8,9,10,11,12,13,14,15);
    #pragma unroll
    for (int s = 0; s < 4; ++s) {
      const v16bf b = *(const v16bf*)(W + (size_t)(nb + (s << 4)) * K + k0 + kboff);
      acc[0][s] = __builtin_amdgcn_wmma_f32_16x16x32_bf16(
          false, a0, false, b, (short)0, acc[0][s], false, false);
      acc[1][s] = __builtin_amdgcn_wmma_f32_16x16x32_bf16(
          false, a1, false, b, (short)0, acc[1][s], false, false);
    }
  }

  // C/D layout: VGPR r -> row r (lanes 0-15) / row r+8 (lanes 16-31), col = lane&15
  const int roff = (lane & 16) ? 8 : 0;
  if (Cb) {
    #pragma unroll
    for (int h = 0; h < 2; ++h) {
      const int rbase = (tM << 5) + (h << 4) + roff;
      #pragma unroll
      for (int s = 0; s < 4; ++s) {
        const int col = (tN << 6) + (s << 4) + (lane & 15);
        const float bv = bias ? bias[col] : 0.0f;
        #pragma unroll
        for (int r = 0; r < 8; ++r)
          Cb[(size_t)(rbase + r) * N + col] = (__bf16)(acc[h][s][r] + bv);
      }
    }
  } else {
    #pragma unroll
    for (int h = 0; h < 2; ++h) {
      const int rbase = (tM << 5) + (h << 4) + roff;
      #pragma unroll
      for (int s = 0; s < 4; ++s) {
        const int col = (tN << 6) + (s << 4) + (lane & 15);
        const float bv = bias ? bias[col] : 0.0f;
        #pragma unroll
        for (int r = 0; r < 8; ++r)
          Cf[(size_t)(rbase + r) * N + col] = acc[h][s][r] + bv;
      }
    }
  }
}

// ---------------------------------------------------------------------------
// f32 -> bf16 conversion (weights once per launch; a few activations)
// ---------------------------------------------------------------------------
__global__ void cvt_bf16_kernel(const float* __restrict__ src,
                                __bf16* __restrict__ dst, int total)
{
  int i = blockIdx.x * blockDim.x + threadIdx.x;
  if (i < total) dst[i] = (__bf16)src[i];
}

// ---------------------------------------------------------------------------
// AdaLN: y = (1 + g) * LN(x) + b, g = ab[:, :128], b = ab[:, 128:].
// Writes bf16 (consumers are GEMMs only). One 128-thread block per row.
// ---------------------------------------------------------------------------
__global__ __launch_bounds__(128) void ln_mod_kernel(
    const float* __restrict__ x, const float* __restrict__ ab,
    const float* __restrict__ g, const float* __restrict__ b,
    __bf16* __restrict__ y, int n)
{
  __shared__ float red[128];
  const int row = blockIdx.x;
  const int t = threadIdx.x;
  const float v = x[(size_t)row * D_ATOM + t];
  red[t] = v;
  __syncthreads();
  for (int s = 64; s > 0; s >>= 1) { if (t < s) red[t] += red[t + s]; __syncthreads(); }
  const float mean = red[0] * (1.0f / D_ATOM);
  __syncthreads();
  const float d = v - mean;
  red[t] = d * d;
  __syncthreads();
  for (int s = 64; s > 0; s >>= 1) { if (t < s) red[t] += red[t + s]; __syncthreads(); }
  const float var = red[0] * (1.0f / D_ATOM);
  float nrm = d * rsqrtf(var + 1e-5f);
  if (g) nrm = nrm * g[t] + b[t];
  const float gg = ab[(size_t)row * (2 * D_ATOM) + t];
  const float bb = ab[(size_t)row * (2 * D_ATOM) + D_ATOM + t];
  y[(size_t)row * D_ATOM + t] = (__bf16)((1.0f + gg) * nrm + bb);
}

// ---------------------------------------------------------------------------
// Pair bias: scatter (p_lm @ w_pb^T + b_pb) into band[h][i][j-i+16] (|i-j|<=16)
// ---------------------------------------------------------------------------
__global__ void zero_kernel(float* __restrict__ p, int n)
{
  int i = blockIdx.x * blockDim.x + threadIdx.x;
  if (i < n) p[i] = 0.0f;
}

__global__ void pair_scatter_kernel(
    const float* __restrict__ p_lm, const float* __restrict__ w_pb,
    const float* __restrict__ b_pb, const int* __restrict__ idx,
    float* __restrict__ band, int P, int n)
{
  const int p = blockIdx.x * blockDim.x + threadIdx.x;
  if (p >= P) return;
  const int i = idx[2 * p], j = idx[2 * p + 1];
  const int d = j - i;
  if (d < -WIN || d > WIN) return;
  float x[16];
  #pragma unroll
  for (int t = 0; t < 16; ++t) x[t] = p_lm[(size_t)p * 16 + t];
  #pragma unroll
  for (int h = 0; h < NHEAD; ++h) {
    float acc = b_pb[h];
    #pragma unroll
    for (int t = 0; t < 16; ++t) acc += x[t] * w_pb[h * 16 + t];
    band[((size_t)h * n + i) * WSPAN + (d + WIN)] = acc;
  }
}

// ---------------------------------------------------------------------------
// Windowed attention: 32 atoms/block, thread = (atom, head).
// K/V halo tiles staged in LDS (64 KB of the 320 KB WGP LDS). Output bf16
// (consumer is the w_o GEMM).
// ---------------------------------------------------------------------------
__global__ __launch_bounds__(128) void attn_kernel(
    const float* __restrict__ Q, const float* __restrict__ Kg,
    const float* __restrict__ Vg, const float* __restrict__ band,
    __bf16* __restrict__ out, int n)
{
  __shared__ float sK[(ATT_BI + 2 * WIN) * D_ATOM];
  __shared__ float sV[(ATT_BI + 2 * WIN) * D_ATOM];
  const int i0 = blockIdx.x * ATT_BI;
  for (int e = threadIdx.x; e < (ATT_BI + 2 * WIN) * D_ATOM; e += 128) {
    const int jr = i0 - WIN + (e >> 7);
    const int c = e & (D_ATOM - 1);
    const bool ok = (jr >= 0) && (jr < n);
    sK[e] = ok ? Kg[(size_t)jr * D_ATOM + c] : 0.0f;
    sV[e] = ok ? Vg[(size_t)jr * D_ATOM + c] : 0.0f;
  }
  __syncthreads();

  const int il = threadIdx.x >> 2;
  const int h  = threadIdx.x & 3;
  const int i  = i0 + il;

  float qv[D_H];
  #pragma unroll
  for (int d = 0; d < D_H; ++d) qv[d] = Q[(size_t)i * D_ATOM + h * D_H + d];
  const float* brow = band + ((size_t)h * n + i) * WSPAN;

  float sc[WSPAN];
  float mx = -1e30f;
  #pragma unroll
  for (int w = 0; w < WSPAN; ++w) {
    const int j = i - WIN + w;
    float s = -1e30f;
    if (j >= 0 && j < n) {
      const float* kr = &sK[(il + w) * D_ATOM + h * D_H];
      float dot = 0.0f;
      #pragma unroll
      for (int d = 0; d < D_H; ++d) dot += qv[d] * kr[d];
      s = dot * 0.17677669529663687f + brow[w];   // 1/sqrt(32)
    }
    sc[w] = s;
    mx = fmaxf(mx, s);
  }
  float sum = 0.0f;
  #pragma unroll
  for (int w = 0; w < WSPAN; ++w) { const float e = __expf(sc[w] - mx); sc[w] = e; sum += e; }
  const float inv = 1.0f / sum;

  float o[D_H];
  #pragma unroll
  for (int d = 0; d < D_H; ++d) o[d] = 0.0f;
  #pragma unroll
  for (int w = 0; w < WSPAN; ++w) {
    const float e = sc[w];
    const float* vr = &sV[(il + w) * D_ATOM + h * D_H];
    #pragma unroll
    for (int d = 0; d < D_H; ++d) o[d] += e * vr[d];
  }
  #pragma unroll
  for (int d = 0; d < D_H; ++d)
    out[(size_t)i * D_ATOM + h * D_H + d] = (__bf16)(o[d] * inv);
}

// ---------------------------------------------------------------------------
// Elementwise glue
// ---------------------------------------------------------------------------
__global__ void vecadd_kernel(const float* __restrict__ a, const float* __restrict__ b,
                              float* __restrict__ c, int n)
{
  int i = blockIdx.x * blockDim.x + threadIdx.x;
  if (i < n) c[i] = a[i] + b[i];
}

// q1 = q + sig(G)*oo ; q2 = q1*(1 + sig(gA))
__global__ void fuse_q2_kernel(const float* __restrict__ q, const float* __restrict__ G,
                               const float* __restrict__ oo, const float* __restrict__ gA,
                               float* __restrict__ q2, int total)
{
  int i = blockIdx.x * blockDim.x + threadIdx.x;
  if (i >= total) return;
  const float q1 = q[i] + sigmoidf_(G[i]) * oo[i];
  q2[i] = q1 * (1.0f + sigmoidf_(gA[i]));
}

// h = silu(s1) * s2, stored bf16 (consumer is the w_s3 GEMM)
__global__ void silu_mul_kernel(const float* __restrict__ a, const float* __restrict__ b,
                                __bf16* __restrict__ c, int total)
{
  int i = blockIdx.x * blockDim.x + threadIdx.x;
  if (i >= total) return;
  const float x = a[i];
  c[i] = (__bf16)((x * sigmoidf_(x)) * b[i]);
}

__global__ void final_kernel(const float* __restrict__ q2, const float* __restrict__ gB,
                             const float* __restrict__ swi, float* __restrict__ out, int total)
{
  int i = blockIdx.x * blockDim.x + threadIdx.x;
  if (i >= total) return;
  out[i] = q2[i] + sigmoidf_(gB[i]) * swi[i];
}

// ---------------------------------------------------------------------------
extern "C" void kernel_launch(void* const* d_in, const int* in_sizes, int n_in,
                              void* d_out, int out_size, void* d_ws, size_t ws_size,
                              hipStream_t stream)
{
  (void)n_in; (void)out_size; (void)ws_size;

  const float* q       = (const float*)d_in[0];
  const float* c_atom  = (const float*)d_in[1];
  const float* h_cond  = (const float*)d_in[2];
  const float* p_lm    = (const float*)d_in[3];
  const float* t_emb   = (const float*)d_in[4];
  const float* w_cond  = (const float*)d_in[5];
  const float* b_cond  = (const float*)d_in[6];
  const float* w_a1    = (const float*)d_in[7];
  const float* b_a1    = (const float*)d_in[8];
  const float* w_a2    = (const float*)d_in[9];
  const float* b_a2    = (const float*)d_in[10];
  const float* ln_g    = (const float*)d_in[11];
  const float* ln_b    = (const float*)d_in[12];
  const float* w_q     = (const float*)d_in[13];
  const float* w_k     = (const float*)d_in[14];
  const float* w_v     = (const float*)d_in[15];
  const float* w_g     = (const float*)d_in[16];
  const float* w_o     = (const float*)d_in[17];
  const float* w_pb    = (const float*)d_in[18];
  const float* b_pb    = (const float*)d_in[19];
  const float* w_g1    = (const float*)d_in[20];
  const float* b_g1    = (const float*)d_in[21];
  const float* w_g2    = (const float*)d_in[22];
  const float* b_g2    = (const float*)d_in[23];
  const float* w_s1    = (const float*)d_in[24];
  const float* w_s2    = (const float*)d_in[25];
  const float* w_s3    = (const float*)d_in[26];
  const int* p_lm_idx  = (const int*)d_in[27];
  const int* token_idx = (const int*)d_in[28];
  float* out = (float*)d_out;

  const int n  = in_sizes[0] / D_ATOM;   // 4096
  const int P  = in_sizes[3] / 16;       // 131072
  const int nh = in_sizes[2];            // h_cond elements (n_tok * 512)

  // ---- workspace carving: f32 region, then bf16 region ----
  float* ws = (float*)d_ws;
  size_t off = 0;
  auto take = [&](size_t cnt) { float* p = ws + off; off += cnt; return p; };
  float* abuf  = take((size_t)n * 2 * D_ATOM);   // a1, later a2 (f32)
  float* Qb    = take((size_t)n * D_ATOM);
  float* Kb    = take((size_t)n * D_ATOM);
  float* Vb    = take((size_t)n * D_ATOM);
  float* Gb    = take((size_t)n * D_ATOM);
  float* band  = take((size_t)NHEAD * n * WSPAN);
  float* oo    = take((size_t)n * D_ATOM);
  float* gbuf  = take((size_t)n * D_ATOM);       // gA, later gB
  float* q2    = take((size_t)n * D_ATOM);
  float* s1    = take((size_t)n * 4 * D_ATOM);
  float* s2    = take((size_t)n * 4 * D_ATOM);
  float* swi   = take((size_t)n * D_ATOM);
  float* biasc = take(D_ATOM);

  __bf16* hws = (__bf16*)(ws + off);
  size_t hoff = 0;
  auto takeh = [&](size_t cnt) { __bf16* p = hws + hoff; hoff += cnt; return p; };
  __bf16* h_condh = takeh((size_t)nh);
  __bf16* c_atomh = takeh((size_t)n * D_ATOM);
  __bf16* condh   = takeh((size_t)n * D_ATOM);
  __bf16* qnh     = takeh((size_t)n * D_ATOM);   // q_n, later q_n2
  __bf16* atth    = takeh((size_t)n * D_ATOM);
  __bf16* hb      = takeh((size_t)n * 4 * D_ATOM);
  __bf16* hw_cond = takeh((size_t)D_ATOM * D_MODEL);
  __bf16* hw_a1   = takeh((size_t)2 * D_ATOM * D_ATOM);
  __bf16* hw_a2   = takeh((size_t)2 * D_ATOM * D_ATOM);
  __bf16* hw_q    = takeh((size_t)D_ATOM * D_ATOM);
  __bf16* hw_k    = takeh((size_t)D_ATOM * D_ATOM);
  __bf16* hw_v    = takeh((size_t)D_ATOM * D_ATOM);
  __bf16* hw_g    = takeh((size_t)D_ATOM * D_ATOM);
  __bf16* hw_o    = takeh((size_t)D_ATOM * D_ATOM);
  __bf16* hw_g1   = takeh((size_t)D_ATOM * D_ATOM);
  __bf16* hw_g2   = takeh((size_t)D_ATOM * D_ATOM);
  __bf16* hw_s1   = takeh((size_t)4 * D_ATOM * D_ATOM);
  __bf16* hw_s2   = takeh((size_t)4 * D_ATOM * D_ATOM);
  __bf16* hw_s3   = takeh((size_t)D_ATOM * 4 * D_ATOM);

  auto cvt = [&](const float* s, __bf16* d, int total) {
    cvt_bf16_kernel<<<(total + 255) / 256, 256, 0, stream>>>(s, d, total);
  };
  auto gemm = [&](const __bf16* A, const int* ridx, const __bf16* W, const float* bias,
                  float* Cf, __bf16* Cb, int M, int N, int K) {
    const int blocks = (M / 32) * (N / 64) / 8;   // 8 waves/block, 32x64 tile/wave
    gemm_bf16_wmma_kernel<<<blocks, 256, 0, stream>>>(A, ridx, W, bias, Cf, Cb, M, N, K);
  };

  // 0) one-time bf16 staging of weights + GEMM-side inputs
  cvt(h_cond, h_condh, nh);
  cvt(c_atom, c_atomh, n * D_ATOM);
  cvt(w_cond, hw_cond, D_ATOM * D_MODEL);
  cvt(w_a1, hw_a1, 2 * D_ATOM * D_ATOM);
  cvt(w_a2, hw_a2, 2 * D_ATOM * D_ATOM);
  cvt(w_q, hw_q, D_ATOM * D_ATOM);
  cvt(w_k, hw_k, D_ATOM * D_ATOM);
  cvt(w_v, hw_v, D_ATOM * D_ATOM);
  cvt(w_g, hw_g, D_ATOM * D_ATOM);
  cvt(w_o, hw_o, D_ATOM * D_ATOM);
  cvt(w_g1, hw_g1, D_ATOM * D_ATOM);
  cvt(w_g2, hw_g2, D_ATOM * D_ATOM);
  cvt(w_s1, hw_s1, 4 * D_ATOM * D_ATOM);
  cvt(w_s2, hw_s2, 4 * D_ATOM * D_ATOM);
  cvt(w_s3, hw_s3, D_ATOM * 4 * D_ATOM);

  // 1) cond = t_emb + gather(h_cond) @ w_cond^T + b_cond   (bf16 out: GEMM-fed)
  vecadd_kernel<<<1, D_ATOM, 0, stream>>>(b_cond, t_emb, biasc, D_ATOM);
  gemm(h_condh, token_idx, hw_cond, biasc, nullptr, condh, n, D_ATOM, D_MODEL);

  // 2) a1, AdaLN -> q_n (bf16)
  gemm(condh, nullptr, hw_a1, b_a1, abuf, nullptr, n, 2 * D_ATOM, D_ATOM);
  ln_mod_kernel<<<n, 128, 0, stream>>>(q, abuf, ln_g, ln_b, qnh, n);

  // 3) Q,K,V,G projections (f32 out: VALU consumers)
  gemm(qnh, nullptr, hw_q, nullptr, Qb, nullptr, n, D_ATOM, D_ATOM);
  gemm(qnh, nullptr, hw_k, nullptr, Kb, nullptr, n, D_ATOM, D_ATOM);
  gemm(qnh, nullptr, hw_v, nullptr, Vb, nullptr, n, D_ATOM, D_ATOM);
  gemm(qnh, nullptr, hw_g, nullptr, Gb, nullptr, n, D_ATOM, D_ATOM);

  // 4) banded pair bias + windowed attention (att bf16: GEMM-fed)
  const int bandN = NHEAD * n * WSPAN;
  zero_kernel<<<(bandN + 255) / 256, 256, 0, stream>>>(band, bandN);
  pair_scatter_kernel<<<(P + 255) / 256, 256, 0, stream>>>(p_lm, w_pb, b_pb, p_lm_idx, band, P, n);
  attn_kernel<<<n / ATT_BI, 128, 0, stream>>>(Qb, Kb, Vb, band, atth, n);

  // 5) output projection + gates -> q2
  gemm(atth, nullptr, hw_o, nullptr, oo, nullptr, n, D_ATOM, D_ATOM);
  gemm(c_atomh, nullptr, hw_g1, b_g1, gbuf, nullptr, n, D_ATOM, D_ATOM);
  const int tot = n * D_ATOM;
  fuse_q2_kernel<<<(tot + 255) / 256, 256, 0, stream>>>(q, Gb, oo, gbuf, q2, tot);

  // 6) second AdaLN + SwiGLU
  gemm(condh, nullptr, hw_a2, b_a2, abuf, nullptr, n, 2 * D_ATOM, D_ATOM);
  ln_mod_kernel<<<n, 128, 0, stream>>>(q2, abuf, nullptr, nullptr, qnh, n);
  gemm(qnh, nullptr, hw_s1, nullptr, s1, nullptr, n, 4 * D_ATOM, D_ATOM);
  gemm(qnh, nullptr, hw_s2, nullptr, s2, nullptr, n, 4 * D_ATOM, D_ATOM);
  const int tot4 = n * 4 * D_ATOM;
  silu_mul_kernel<<<(tot4 + 255) / 256, 256, 0, stream>>>(s1, s2, hb, tot4);
  gemm(hb, nullptr, hw_s3, nullptr, swi, nullptr, n, D_ATOM, 4 * D_ATOM);

  // 7) final gate + residual
  gemm(c_atomh, nullptr, hw_g2, b_g2, gbuf, nullptr, n, D_ATOM, D_ATOM);
  final_kernel<<<(tot + 255) / 256, 256, 0, stream>>>(q2, gbuf, swi, out, tot);
}
